// CAM_GRU_79293686219199
// MI455X (gfx1250) — compile-verified, matrix operand order checked by reading, not run on hardware
//
#include <hip/hip_runtime.h>
#include <math.h>

// ---------------------------------------------------------------------------
// CDNA5 (gfx1250) implementation: f16 WMMA (16x16x32, f32 accum) for all
// matmuls; persistent single-workgroup recurrences with LDS-resident state.
// ---------------------------------------------------------------------------

typedef __attribute__((ext_vector_type(16))) _Float16 v16h;
typedef __attribute__((ext_vector_type(8)))  _Float16 v8h;
typedef __attribute__((ext_vector_type(8)))  float    v8f;

#define B_   32
#define S_   1000
#define T_   184
#define H_   500
#define HP   512     // padded H (K multiple of 32)
#define G3   1500    // 3*H
#define G3P  1536    // padded gate stride == 24 groups * 64
#define NP3  1536    // padded weight rows for N=1500 (multiple of 64)
#define FE_  184
#define FEP  192
#define H2P  1024    // padded 2H
#define K1   552     // 3*T
#define K1P  576
#define N1   368
#define LD1  384     // = 6 groups * 64
#define N2   184
#define LD2  192     // = 3 groups * 64

// ---------------------------------------------------------------------------
// WMMA tile helper: one wave computes a 16x16 f32 tile of A(16xK) * W(16xK)^T.
// A/W row-major f16 with K padded to a multiple of 32 (zero padded).
// Per-lane fragment addressing follows the CDNA5 16-bit A/B layout:
//   row = lane%16 ; halves [0..7] at k0+( (lane>=16)?8:0 ), halves [8..15] at +16.
// ---------------------------------------------------------------------------
__device__ inline v8f wtile(const _Float16* A, int lda,
                            const _Float16* Wt, int ldw, int K) {
  int lane = threadIdx.x & 31;
  const _Float16* ap = A  + (size_t)(lane & 15) * lda + ((lane >> 4) << 3);
  const _Float16* wp = Wt + (size_t)(lane & 15) * ldw + ((lane >> 4) << 3);
  v8f acc = {};
  for (int k = 0; k < K; k += 32) {
    v16h a, b;
    ((v8h*)&a)[0] = *(const v8h*)(ap + k);
    ((v8h*)&a)[1] = *(const v8h*)(ap + k + 16);
    ((v8h*)&b)[0] = *(const v8h*)(wp + k);
    ((v8h*)&b)[1] = *(const v8h*)(wp + k + 16);
    acc = __builtin_amdgcn_wmma_f32_16x16x32_f16(false, a, false, b,
                                                 (short)0, acc, false, false);
  }
  return acc;
}

// ---------------------------------------------------------------------------
// Generic tiled GEMM: C[M,N] = A[M,K] @ W[N,K]^T + bias.
// Register-blocked 1x4 with NO tail: weight rows are padded to ngroups*64 and
// every output leading dimension is >= ngroups*64 columns, so each wave
// unconditionally computes 4 adjacent N tiles (A fragment loaded once per
// K-step, reused across 4 v_wmma; no EXEC manipulation anywhere).
// Padded columns evaluate to 0 (zero weight rows + guarded bias), which is
// exactly what downstream padded-K GEMMs require.
// ---------------------------------------------------------------------------
__global__ __launch_bounds__(256)
void k_gemm(const _Float16* __restrict__ A, int lda,
            const _Float16* __restrict__ W, int ldw,
            const float* __restrict__ bias,
            float* __restrict__ C, int ldc,
            _Float16* __restrict__ Ch, int ldch,
            int M, int N, int K)
{
  int ntiles  = (N + 15) >> 4;
  int mtiles  = M >> 4;
  int ngroups = (ntiles + 3) >> 2;
  int wv = (int)((blockIdx.x * blockDim.x + threadIdx.x) >> 5);
  if (wv >= mtiles * ngroups) return;        // wave-uniform; EXEC all-1 inside
  int mt = wv / ngroups, ng = wv - mt * ngroups;
  int nt0 = ng << 2;
  int lane = threadIdx.x & 31;
  int khi = (lane >> 4) << 3;
  const _Float16* ap  = A + (size_t)(mt * 16 + (lane & 15)) * lda + khi;
  const _Float16* wp0 = W + (size_t)(nt0 * 16 + (lane & 15)) * ldw + khi;
  if (K >= 256) __builtin_prefetch(wp0, 0, 1);
  v8f acc0 = {}, acc1 = {}, acc2 = {}, acc3 = {};
  for (int k = 0; k < K; k += 32) {
    v16h a, b0, b1, b2, b3;
    ((v8h*)&a)[0]  = *(const v8h*)(ap + k);
    ((v8h*)&a)[1]  = *(const v8h*)(ap + k + 16);
    ((v8h*)&b0)[0] = *(const v8h*)(wp0 + k);
    ((v8h*)&b0)[1] = *(const v8h*)(wp0 + k + 16);
    ((v8h*)&b1)[0] = *(const v8h*)(wp0 + (size_t)16 * ldw + k);
    ((v8h*)&b1)[1] = *(const v8h*)(wp0 + (size_t)16 * ldw + k + 16);
    ((v8h*)&b2)[0] = *(const v8h*)(wp0 + (size_t)32 * ldw + k);
    ((v8h*)&b2)[1] = *(const v8h*)(wp0 + (size_t)32 * ldw + k + 16);
    ((v8h*)&b3)[0] = *(const v8h*)(wp0 + (size_t)48 * ldw + k);
    ((v8h*)&b3)[1] = *(const v8h*)(wp0 + (size_t)48 * ldw + k + 16);
    acc0 = __builtin_amdgcn_wmma_f32_16x16x32_f16(false, a, false, b0,
                                                  (short)0, acc0, false, false);
    acc1 = __builtin_amdgcn_wmma_f32_16x16x32_f16(false, a, false, b1,
                                                  (short)0, acc1, false, false);
    acc2 = __builtin_amdgcn_wmma_f32_16x16x32_f16(false, a, false, b2,
                                                  (short)0, acc2, false, false);
    acc3 = __builtin_amdgcn_wmma_f32_16x16x32_f16(false, a, false, b3,
                                                  (short)0, acc3, false, false);
  }
  int r0 = mt * 16 + ((lane >> 4) << 3);
  v8f accs[4] = {acc0, acc1, acc2, acc3};
#pragma unroll
  for (int j = 0; j < 4; ++j) {
    int n = (nt0 + j) * 16 + (lane & 15);
    float bs = (bias && n < N) ? bias[n] : 0.0f;
#pragma unroll
    for (int v = 0; v < 8; ++v) {
      float x = accs[j][v] + bs;             // padded cols: acc==0,bias==0 -> 0
      size_t row = (size_t)(r0 + v);
      if (C)  C [row * (size_t)ldc  + n] = x;
      if (Ch) Ch[row * (size_t)ldch + n] = (_Float16)x;
    }
  }
}

// ---------------------------------------------------------------------------
// Persistent encoder GRU (one direction, one layer). Single workgroup of
// 32 waves. Hidden state f16 in LDS. Per step: gh = h @ Whh^T via WMMA
// (2x94 tiles over 32 waves), then fused gate math.  gi (input projection,
// bih already added) was precomputed by k_gemm.
// ---------------------------------------------------------------------------
__global__ __launch_bounds__(1024)
void k_gru_seq(const float* __restrict__ gi,          // (B*S) x G3P
               const _Float16* __restrict__ Whh,      // NP3 x HP (zero-padded)
               const float* __restrict__ bhh,         // 1500
               _Float16* __restrict__ y, int ldy,     // dest (pre-offset)
               float* __restrict__ gh,                // scratch B_ x G3P
               float* __restrict__ hT,                // (B,H) or null
               int S, int reverse)
{
  __shared__ __align__(16) _Float16 hh[B_][HP];
  int tid = threadIdx.x, lane = tid & 31, wv = tid >> 5;
  for (int i = tid; i < B_ * HP; i += 1024) ((_Float16*)hh)[i] = (_Float16)0.f;
  __syncthreads();
  const int ntiles = 94, tiles = 2 * ntiles;
  for (int t = 0; t < S; ++t) {
    int s = reverse ? (S - 1 - t) : t;
    for (int tt = wv; tt < tiles; tt += 32) {
      int mt = tt / ntiles, nt = tt - mt * ntiles;
      v8f acc = wtile(&hh[mt * 16][0], HP, Whh + (size_t)(nt * 16) * HP, HP, HP);
      int n  = nt * 16 + (lane & 15);
      int r0 = mt * 16 + ((lane >> 4) << 3);
      for (int v = 0; v < 8; ++v) gh[(size_t)(r0 + v) * G3P + n] = acc[v];
    }
    __syncthreads();
    for (int idx = tid; idx < B_ * H_; idx += 1024) {
      int b = idx / H_, j = idx - b * H_;
      const float* gir = gi + ((size_t)b * S + s) * G3P;
      float hr = gh[(size_t)b * G3P + j]          + bhh[j];
      float hz = gh[(size_t)b * G3P + j + H_]     + bhh[j + H_];
      float hn = gh[(size_t)b * G3P + j + 2 * H_] + bhh[j + 2 * H_];
      float r = 1.f / (1.f + __expf(-(gir[j] + hr)));
      float z = 1.f / (1.f + __expf(-(gir[j + H_] + hz)));
      float n = tanhf(gir[j + 2 * H_] + r * hn);
      float h = (1.f - z) * n + z * (float)hh[b][j];
      _Float16 h16 = (_Float16)h;
      hh[b][j] = h16;
      y[((size_t)b * S + s) * (size_t)ldy + j] = h16;
    }
    __syncthreads();
  }
  if (hT)
    for (int idx = tid; idx < B_ * H_; idx += 1024)
      hT[idx] = (float)hh[idx / H_][idx % H_];
}

// Shared GRU gate update for the decoder (state f16 in LDS, gates in global).
__device__ inline void dec_gates(const float* gi_, const float* gh_,
                                 const float* bih, const float* bhh,
                                 _Float16 (*hh)[HP])
{
  for (int idx = threadIdx.x; idx < 16 * H_; idx += 1024) {
    int b = idx / H_, j = idx - b * H_;
    float ir = gi_[(size_t)b * G3P + j]          + bih[j];
    float iz = gi_[(size_t)b * G3P + j + H_]     + bih[j + H_];
    float in_= gi_[(size_t)b * G3P + j + 2 * H_] + bih[j + 2 * H_];
    float hr = gh_[(size_t)b * G3P + j]          + bhh[j];
    float hz = gh_[(size_t)b * G3P + j + H_]     + bhh[j + H_];
    float hn = gh_[(size_t)b * G3P + j + 2 * H_] + bhh[j + 2 * H_];
    float r = 1.f / (1.f + __expf(-(ir + hr)));
    float z = 1.f / (1.f + __expf(-(iz + hz)));
    float n = tanhf(in_ + r * hn);
    float h = (1.f - z) * n + z * (float)hh[b][j];
    hh[b][j] = (_Float16)h;
  }
}

// ---------------------------------------------------------------------------
// Decoder: 2 workgroups x 16 batch rows, persistent over T=184 steps.
// GRU cell / q / attn projections via WMMA (M=16 tiles). Attention dot
// products + softmax with VALU + wave shuffles; enc_keyed (64MB) is
// L2-resident on MI455X (192MB L2).
// ---------------------------------------------------------------------------
__global__ __launch_bounds__(1024)
void k_decoder(const _Float16* __restrict__ yf2, const _Float16* __restrict__ yb2,
               const float* __restrict__ enc_inp,
               const float* __restrict__ hTf, const float* __restrict__ hTb,
               const _Float16* __restrict__ Wih0, const float* __restrict__ bih0,
               const _Float16* __restrict__ Whh0, const float* __restrict__ bhh0,
               const _Float16* __restrict__ Wih1, const float* __restrict__ bih1,
               const _Float16* __restrict__ Whh1, const float* __restrict__ bhh1,
               const _Float16* __restrict__ qW,   const float* __restrict__ qb,
               const float* __restrict__ keyed,   // (B*S) x HP f32, pad cols 0
               const _Float16* __restrict__ attnW,const float* __restrict__ attnb,
               float* __restrict__ Y0,            // (B*H) x K1P, cols [0,T)
               float* __restrict__ hout,          // (2,B,H)
               char* __restrict__ scr, size_t wgbytes)
{
  __shared__ __align__(16) _Float16 hh0[16][HP], hh1[16][HP], xin[16][HP];
  __shared__ float red[16][2];
  int tid = threadIdx.x, lane = tid & 31, wv = tid >> 5;
  int bm = blockIdx.x * 16;
  char* base = scr + (size_t)blockIdx.x * wgbytes;
  float*    g_gi  = (float*)base;
  float*    g_gh  = g_gi + 16 * G3P;
  float*    g_sc  = g_gh + 16 * G3P;
  _Float16* g_q   = (_Float16*)(g_sc + 16 * 1024);
  _Float16* g_cat = g_q + 16 * HP;

  for (int i = tid; i < 16 * HP; i += 1024) {
    int b = i >> 9, j = i & (HP - 1);
    float v0 = (j < H_) ? hTf[(size_t)(bm + b) * H_ + j] : 0.f;
    float v1 = (j < H_) ? hTb[(size_t)(bm + b) * H_ + j] : 0.f;
    hh0[b][j] = (_Float16)v0;
    hh1[b][j] = (_Float16)v1;
    float xv;
    if (j < H_)
      xv = (float)yf2[((size_t)(bm + b) * S_ + (S_ - 1)) * HP + j]
         + (float)yb2[((size_t)(bm + b) * S_ + (S_ - 1)) * HP + j];
    else
      xv = enc_inp[((size_t)(bm + b) * T_) * 12 + (j - H_)];
    xin[b][j] = (_Float16)xv;
    g_q[i] = (_Float16)0.f;
  }
  for (int i = tid; i < 16 * H2P; i += 1024) g_cat[i] = (_Float16)0.f;
  __syncthreads();

  for (int t = 0; t < T_; ++t) {
    // ---- layer 0: gi = xin @ Wih0^T ; gh = h0 @ Whh0^T  (94 tiles each) ----
    for (int tt = wv; tt < 94; tt += 32) {
      v8f acc = wtile(&xin[0][0], HP, Wih0 + (size_t)(tt * 16) * HP, HP, HP);
      int n = tt * 16 + (lane & 15), r0 = (lane >> 4) << 3;
      for (int v = 0; v < 8; ++v) g_gi[(size_t)(r0 + v) * G3P + n] = acc[v];
    }
    for (int tt = wv; tt < 94; tt += 32) {
      v8f acc = wtile(&hh0[0][0], HP, Whh0 + (size_t)(tt * 16) * HP, HP, HP);
      int n = tt * 16 + (lane & 15), r0 = (lane >> 4) << 3;
      for (int v = 0; v < 8; ++v) g_gh[(size_t)(r0 + v) * G3P + n] = acc[v];
    }
    __syncthreads();
    dec_gates(g_gi, g_gh, bih0, bhh0, hh0);
    __syncthreads();
    // ---- layer 1 ----
    for (int tt = wv; tt < 94; tt += 32) {
      v8f acc = wtile(&hh0[0][0], HP, Wih1 + (size_t)(tt * 16) * HP, HP, HP);
      int n = tt * 16 + (lane & 15), r0 = (lane >> 4) << 3;
      for (int v = 0; v < 8; ++v) g_gi[(size_t)(r0 + v) * G3P + n] = acc[v];
    }
    for (int tt = wv; tt < 94; tt += 32) {
      v8f acc = wtile(&hh1[0][0], HP, Whh1 + (size_t)(tt * 16) * HP, HP, HP);
      int n = tt * 16 + (lane & 15), r0 = (lane >> 4) << 3;
      for (int v = 0; v < 8; ++v) g_gh[(size_t)(r0 + v) * G3P + n] = acc[v];
    }
    __syncthreads();
    // layer-1 gates + routing of temp into cat / (t==0) direct output
    for (int idx = tid; idx < 16 * H_; idx += 1024) {
      int b = idx / H_, j = idx - b * H_;
      float ir = g_gi[(size_t)b * G3P + j]          + bih1[j];
      float iz = g_gi[(size_t)b * G3P + j + H_]     + bih1[j + H_];
      float in_= g_gi[(size_t)b * G3P + j + 2 * H_] + bih1[j + 2 * H_];
      float hr = g_gh[(size_t)b * G3P + j]          + bhh1[j];
      float hz = g_gh[(size_t)b * G3P + j + H_]     + bhh1[j + H_];
      float hn = g_gh[(size_t)b * G3P + j + 2 * H_] + bhh1[j + 2 * H_];
      float r = 1.f / (1.f + __expf(-(ir + hr)));
      float z = 1.f / (1.f + __expf(-(iz + hz)));
      float n = tanhf(in_ + r * hn);
      float h = (1.f - z) * n + z * (float)hh1[b][j];
      _Float16 h16 = (_Float16)h;
      hh1[b][j] = h16;
      g_cat[(size_t)b * H2P + HP + j] = h16;       // temp half of [ctx,temp]
      if (t == 0) {
        xin[b][j] = h16;                            // prev = out0
        Y0[((size_t)(bm + b) * H_ + j) * K1P] = h;  // column 0
      }
    }
    __syncthreads();

    if (t > 0) {
      // q = temp @ qW^T + qb : exactly one 16x16 tile per wave (32 tiles)
      {
        int tt = wv;
        v8f acc = wtile(&hh1[0][0], HP, qW + (size_t)(tt * 16) * HP, HP, HP);
        int n = tt * 16 + (lane & 15), r0 = (lane >> 4) << 3;
        float bs = (n < H_) ? qb[n] : 0.f;
        for (int v = 0; v < 8; ++v)
          g_q[(size_t)(r0 + v) * HP + n] = (_Float16)(acc[v] + bs);
      }
      __syncthreads();
      // scores[b][s] = q[b] . keyed[b][s]  (pads are zero on both sides)
      {
        int b = wv & 15, sw = wv >> 4;
        const float* kb = keyed + ((size_t)(bm + b) * S_) * HP;
        const _Float16* qrow = g_q + (size_t)b * HP;
        for (int s = sw * 32 + lane; s < S_; s += 64) {
          const float* kp = kb + (size_t)s * HP;
          float d = 0.f;
          for (int j = 0; j < HP; ++j) d += (float)qrow[j] * kp[j];
          g_sc[(size_t)b * 1024 + s] = d;
        }
      }
      __syncthreads();
      // softmax stats: wave w < 16 reduces batch row w (wave32 shuffles)
      if (wv < 16) {
        int b = wv;
        float m = -1e30f;
        for (int s = lane; s < S_; s += 32) m = fmaxf(m, g_sc[(size_t)b * 1024 + s]);
        for (int o = 16; o; o >>= 1) m = fmaxf(m, __shfl_xor(m, o));
        float sum = 0.f;
        for (int s = lane; s < S_; s += 32) sum += __expf(g_sc[(size_t)b * 1024 + s] - m);
        for (int o = 16; o; o >>= 1) sum += __shfl_xor(sum, o);
        if (lane == 0) { red[b][0] = m; red[b][1] = 1.f / sum; }
      }
      __syncthreads();
      for (int i = tid; i < 16 * S_; i += 1024) {
        int b = i / S_, s = i - b * S_;
        g_sc[(size_t)b * 1024 + s] =
            __expf(g_sc[(size_t)b * 1024 + s] - red[b][0]) * red[b][1];
      }
      __syncthreads();
      // ctx[b][j] = sum_s a[b][s] * keyed[b][s][j]
      for (int i = tid; i < 16 * H_; i += 1024) {
        int b = i / H_, j = i - b * H_;
        const float* kp = keyed + ((size_t)(bm + b) * S_) * HP + j;
        const float* ap = g_sc + (size_t)b * 1024;
        float acc = 0.f;
        for (int s = 0; s < S_; ++s) acc += ap[s] * kp[(size_t)s * HP];
        g_cat[(size_t)b * H2P + j] = (_Float16)acc;
      }
      __syncthreads();
      // o = tanh([ctx,temp] @ attnW^T + attnb) -> Y0 col t, prev for next step
      {
        int tt = wv;
        v8f acc = wtile(g_cat, H2P, attnW + (size_t)(tt * 16) * H2P, H2P, H2P);
        int n = tt * 16 + (lane & 15), r0 = (lane >> 4) << 3;
        float bs = (n < H_) ? attnb[n] : 0.f;
        for (int v = 0; v < 8; ++v) {
          float x = tanhf(acc[v] + bs);
          if (n < H_) {
            xin[r0 + v][n] = (_Float16)x;
            Y0[((size_t)(bm + r0 + v) * H_ + n) * K1P + t] = x;
          }
        }
      }
      __syncthreads();
    }
    // stage e_{t+1} into xin[:, 500..511]
    if (t + 1 < T_) {
      for (int i = tid; i < 16 * 12; i += 1024) {
        int b = i / 12, f = i - b * 12;
        xin[b][H_ + f] =
            (_Float16)enc_inp[((size_t)(bm + b) * T_ + (t + 1)) * 12 + f];
      }
    }
    __syncthreads();
  }
  for (int i = tid; i < 16 * H_; i += 1024) {
    int b = i / H_, j = i - b * H_;
    hout[(size_t)(bm + b) * H_ + j]                      = (float)hh0[b][j];
    hout[(size_t)B_ * H_ + (size_t)(bm + b) * H_ + j]    = (float)hh1[b][j];
  }
}

// ---------------------------------------------------------------------------
// Utility kernels: f32 -> padded f16 repack (with optional column split for
// concat layouts), zero-fill, bidirectional sum, pos-emb head, Y0 fill,
// final transpose.
// ---------------------------------------------------------------------------
__global__ void k_pack(_Float16* __restrict__ dst, const float* __restrict__ src,
                       long long drows, int dcols, long long srows, int scols,
                       int seg1, int dseg2)
{
  long long i = (long long)blockIdx.x * blockDim.x + threadIdx.x;
  if (i >= drows * (long long)dcols) return;
  long long r = i / dcols; int dc = (int)(i - r * dcols);
  float v = 0.f;
  if (r < srows) {
    int sc = -1;
    if (dc < seg1) sc = dc;
    else if (dc >= dseg2 && dc < dseg2 + (scols - seg1)) sc = seg1 + (dc - dseg2);
    if (sc >= 0 && sc < scols) v = src[r * (long long)scols + sc];
  }
  dst[i] = (_Float16)v;
}

__global__ void k_zero16(_Float16* p, long long n) {
  long long i = (long long)blockIdx.x * blockDim.x + threadIdx.x;
  if (i < n) p[i] = (_Float16)0.f;
}

__global__ void k_esum(_Float16* __restrict__ d, const _Float16* __restrict__ a,
                       const _Float16* __restrict__ b, long long n) {
  long long i = (long long)blockIdx.x * blockDim.x + threadIdx.x;
  if (i >= n) return;
  int col = (int)(i & (HP - 1));
  d[i] = (_Float16)((col < H_) ? ((float)a[i] + (float)b[i]) : 0.f);
}

__global__ void k_pos(float* __restrict__ pos, const float* __restrict__ embW,
                      const float* __restrict__ embb) {
  int i = blockIdx.x * blockDim.x + threadIdx.x;
  if (i >= H_ * T_) return;
  int hd = i / T_, t = i - hd * T_;
  float tv = ((hd < H_ / 2) ? (float)(H_ / 2 - 1 - hd) : (float)(hd - H_ / 2)) * 1e-9f;
  float acc = embb[t];
  for (int d = 0; d < 20; ++d)
    acc += __cosf(tv * (0.5f * (float)d) * (100.0f * 3.14159265358979f))
           * embW[t * 20 + d];
  pos[i] = acc;
}

__global__ void k_y0_fill(float* __restrict__ Y0, const float* __restrict__ xc,
                          const float* __restrict__ pos) {
  long long i = (long long)blockIdx.x * blockDim.x + threadIdx.x;
  const int W = K1P - T_;                         // cols [T_, K1P)
  if (i >= (long long)B_ * H_ * W) return;
  long long row = i / W; int c2 = (int)(i - row * W);
  int b = (int)(row / H_), hd = (int)(row - (long long)b * H_);
  float v;
  if (c2 < T_)            v = xc[(size_t)b * T_ + c2];
  else if (c2 < 2 * T_)   v = pos[(size_t)hd * T_ + (c2 - T_)];
  else                    v = 0.f;
  Y0[row * K1P + (T_ + c2)] = v;
}

__global__ void k_out_y(float* __restrict__ out, const float* __restrict__ Y3) {
  long long i = (long long)blockIdx.x * blockDim.x + threadIdx.x;
  if (i >= (long long)B_ * T_ * H_) return;
  int hd = (int)(i % H_);
  long long r = i / H_;
  int t = (int)(r % T_);
  int b = (int)(r / T_);
  out[i] = Y3[((size_t)b * H_ + hd) * LD2 + t];
}

// ---------------------------------------------------------------------------
// Host orchestration
// ---------------------------------------------------------------------------
extern "C" void kernel_launch(void* const* d_in, const int* in_sizes, int n_in,
                              void* d_out, int out_size, void* d_ws, size_t ws_size,
                              hipStream_t stream) {
  (void)in_sizes; (void)n_in; (void)out_size; (void)ws_size;

  const float* in_input = (const float*)d_in[0];   // (32,1000,184)
  const float* in_enc   = (const float*)d_in[1];   // (32,184,12)
  const float* in_xc    = (const float*)d_in[2];   // (32,184)
  const float* g1[4][4];                            // l0f,l0b,l1f,l1b
  for (int i = 0; i < 4; ++i)
    for (int p = 0; p < 4; ++p) g1[i][p] = (const float*)d_in[4 + i * 4 + p];
  const float* g2[2][4];
  for (int i = 0; i < 2; ++i)
    for (int p = 0; p < 4; ++p) g2[i][p] = (const float*)d_in[20 + i * 4 + p];
  const float* keyW_s = (const float*)d_in[28]; const float* key_b = (const float*)d_in[29];
  const float* qW_s   = (const float*)d_in[30]; const float* q_b   = (const float*)d_in[31];
  const float* atW_s  = (const float*)d_in[32]; const float* at_b  = (const float*)d_in[33];
  const float* embW   = (const float*)d_in[34]; const float* emb_b = (const float*)d_in[35];
  const float* o1W    = (const float*)d_in[36]; const float* o1b   = (const float*)d_in[37];
  const float* o2W    = (const float*)d_in[38]; const float* o2b   = (const float*)d_in[39];
  const float* o3W    = (const float*)d_in[40]; const float* o3b   = (const float*)d_in[41];

  // --- workspace carving -------------------------------------------------
  char* ws = (char*)d_ws;
  size_t off = 0;
  auto ALLOC = [&](size_t bytes) -> void* {
    size_t cur = off;
    off = (cur + bytes + 255) & ~(size_t)255;
    return (void*)(ws + cur);
  };
  _Float16* wWhh1[4]; for (int i = 0; i < 4; ++i) wWhh1[i] = (_Float16*)ALLOC((size_t)NP3 * HP * 2);
  _Float16* wWi0[2];  for (int i = 0; i < 2; ++i) wWi0[i]  = (_Float16*)ALLOC((size_t)NP3 * FEP * 2);
  _Float16* wWi1[2];  for (int i = 0; i < 2; ++i) wWi1[i]  = (_Float16*)ALLOC((size_t)NP3 * H2P * 2);
  _Float16* w2ih0 = (_Float16*)ALLOC((size_t)NP3 * HP * 2);
  _Float16* w2hh0 = (_Float16*)ALLOC((size_t)NP3 * HP * 2);
  _Float16* w2ih1 = (_Float16*)ALLOC((size_t)NP3 * HP * 2);
  _Float16* w2hh1 = (_Float16*)ALLOC((size_t)NP3 * HP * 2);
  _Float16* wKey  = (_Float16*)ALLOC((size_t)HP * HP * 2);
  _Float16* wQ    = (_Float16*)ALLOC((size_t)HP * HP * 2);
  _Float16* wAttn = (_Float16*)ALLOC((size_t)HP * H2P * 2);
  _Float16* wO1   = (_Float16*)ALLOC((size_t)LD1 * K1P * 2);   // 384 rows
  _Float16* wO2   = (_Float16*)ALLOC((size_t)LD2 * LD1 * 2);   // 192 rows
  _Float16* wO3   = (_Float16*)ALLOC((size_t)LD2 * LD2 * 2);   // 192 rows
  const size_t MS = (size_t)B_ * S_;               // 32000 rows
  _Float16* x0h   = (_Float16*)ALLOC(MS * FEP * 2);
  _Float16* x1h   = (_Float16*)ALLOC(MS * H2P * 2);
  _Float16* yf2h  = (_Float16*)ALLOC(MS * HP * 2);
  _Float16* yb2h  = (_Float16*)ALLOC(MS * HP * 2);
  _Float16* esumh = (_Float16*)ALLOC(MS * HP * 2);
  _Float16* Y0h   = (_Float16*)ALLOC((size_t)B_ * H_ * K1P * 2);
  _Float16* Y1h   = (_Float16*)ALLOC((size_t)B_ * H_ * LD1 * 2);
  _Float16* Y2h   = (_Float16*)ALLOC((size_t)B_ * H_ * LD2 * 2);
  float* gi    = (float*)ALLOC(MS * G3P * 4);      // reused per layer/dir
  float* ghE   = (float*)ALLOC((size_t)B_ * G3P * 4);
  float* hTf   = (float*)ALLOC((size_t)B_ * H_ * 4);
  float* hTb   = (float*)ALLOC((size_t)B_ * H_ * 4);
  float* keyed = (float*)ALLOC(MS * HP * 4);
  const size_t WGB = (size_t)(2 * 16 * G3P) * 4 + (size_t)16 * 1024 * 4
                   + (size_t)16 * HP * 2 + (size_t)16 * H2P * 2;
  char*  decScr = (char*)ALLOC(2 * WGB);
  float* pos   = (float*)ALLOC((size_t)H_ * T_ * 4);
  float* Y0    = (float*)ALLOC((size_t)B_ * H_ * K1P * 4);
  float* Y3    = (float*)ALLOC((size_t)B_ * H_ * LD2 * 4);

  float* y_out = (float*)d_out;
  float* h_out = y_out + (size_t)B_ * T_ * H_;

  auto PK = [&](_Float16* dst, const float* src, long long dr, int dc,
                long long sr, int sc, int seg1, int dseg2) {
    long long n = dr * (long long)dc;
    k_pack<<<dim3((unsigned)((n + 255) / 256)), dim3(256), 0, stream>>>(
        dst, src, dr, dc, sr, sc, seg1, dseg2);
  };
  auto GEMM = [&](const _Float16* A, int lda, const _Float16* W, int ldw,
                  const float* bias, float* C, int ldc, _Float16* Ch, int ldch,
                  int M, int N, int K) {
    int ntiles = (N + 15) >> 4;
    int groups = (M >> 4) * ((ntiles + 3) >> 2);   // 1x4 register blocking
    k_gemm<<<dim3((unsigned)((groups + 7) / 8)), dim3(256), 0, stream>>>(
        A, lda, W, ldw, bias, C, ldc, Ch, ldch, M, N, K);
  };

  // --- weight repacks (f32 -> padded f16) --------------------------------
  for (int i = 0; i < 4; ++i) PK(wWhh1[i], g1[i][1], NP3, HP, G3, H_, H_, HP);
  for (int d = 0; d < 2; ++d) PK(wWi0[d], g1[d][0], NP3, FEP, G3, FE_, FE_, FEP);
  for (int d = 0; d < 2; ++d) PK(wWi1[d], g1[2 + d][0], NP3, H2P, G3, 2 * H_, H_, HP);
  PK(w2ih0, g2[0][0], NP3, HP, G3, 512, 512, HP);
  PK(w2hh0, g2[0][1], NP3, HP, G3, H_, H_, HP);
  PK(w2ih1, g2[1][0], NP3, HP, G3, H_, H_, HP);
  PK(w2hh1, g2[1][1], NP3, HP, G3, H_, H_, HP);
  PK(wKey, keyW_s, HP, HP, H_, H_, H_, HP);
  PK(wQ,   qW_s,   HP, HP, H_, H_, H_, HP);
  PK(wAttn, atW_s, HP, H2P, H_, 2 * H_, H_, HP);
  PK(wO1, o1W, LD1, K1P, N1, K1, K1, K1P);
  PK(wO2, o2W, LD2, LD1, N2, N1, N1, LD1);
  PK(wO3, o3W, LD2, LD2, N2, N2, N2, LD2);
  PK(x0h, in_input, (long long)MS, FEP, (long long)MS, FE_, FE_, FEP);
  {
    long long n = (long long)MS * H2P;
    k_zero16<<<dim3((unsigned)((n + 255) / 256)), dim3(256), 0, stream>>>(x1h, n);
  }
  k_pos<<<dim3((H_ * T_ + 255) / 256), dim3(256), 0, stream>>>(pos, embW, emb_b);

  // --- encoder: 2-layer bidirectional GRU --------------------------------
  // layer 0 forward
  GEMM(x0h, FEP, wWi0[0], FEP, g1[0][2], gi, G3P, nullptr, 0, (int)MS, G3, FEP);
  k_gru_seq<<<dim3(1), dim3(1024), 0, stream>>>(gi, wWhh1[0], g1[0][3],
      x1h + 0, H2P, ghE, nullptr, S_, 0);
  // layer 0 backward
  GEMM(x0h, FEP, wWi0[1], FEP, g1[1][2], gi, G3P, nullptr, 0, (int)MS, G3, FEP);
  k_gru_seq<<<dim3(1), dim3(1024), 0, stream>>>(gi, wWhh1[1], g1[1][3],
      x1h + HP, H2P, ghE, nullptr, S_, 1);
  // layer 1 forward
  GEMM(x1h, H2P, wWi1[0], H2P, g1[2][2], gi, G3P, nullptr, 0, (int)MS, G3, H2P);
  k_gru_seq<<<dim3(1), dim3(1024), 0, stream>>>(gi, wWhh1[2], g1[2][3],
      yf2h, HP, ghE, hTf, S_, 0);
  // layer 1 backward
  GEMM(x1h, H2P, wWi1[1], H2P, g1[3][2], gi, G3P, nullptr, 0, (int)MS, G3, H2P);
  k_gru_seq<<<dim3(1), dim3(1024), 0, stream>>>(gi, wWhh1[3], g1[3][3],
      yb2h, HP, ghE, hTb, S_, 1);

  // enc_sum = yf + yb ; enc_keyed = enc_sum @ key_W^T + key_b
  {
    long long n = (long long)MS * HP;
    k_esum<<<dim3((unsigned)((n + 255) / 256)), dim3(256), 0, stream>>>(esumh, yf2h, yb2h, n);
  }
  GEMM(esumh, HP, wKey, HP, key_b, keyed, HP, nullptr, 0, (int)MS, H_, HP);

  // Y0 static columns (xcorr + positional), decoder fills cols [0,T)
  {
    long long n = (long long)B_ * H_ * (K1P - T_);
    k_y0_fill<<<dim3((unsigned)((n + 255) / 256)), dim3(256), 0, stream>>>(Y0, in_xc, pos);
  }

  // --- decoder with attention --------------------------------------------
  k_decoder<<<dim3(2), dim3(1024), 0, stream>>>(
      yf2h, yb2h, in_enc, hTf, hTb,
      w2ih0, g2[0][2], w2hh0, g2[0][3],
      w2ih1, g2[1][2], w2hh1, g2[1][3],
      wQ, q_b, keyed, wAttn, at_b,
      Y0, h_out, decScr, WGB);

  // --- output head: 3 chained GEMMs, then transpose to (B,T,H) ------------
  PK(Y0h, Y0, (long long)B_ * H_, K1P, (long long)B_ * H_, K1P, K1P, K1P);
  GEMM(Y0h, K1P, wO1, K1P, o1b, nullptr, 0, Y1h, LD1, B_ * H_, N1, K1P);
  GEMM(Y1h, LD1, wO2, LD1, o2b, nullptr, 0, Y2h, LD2, B_ * H_, N2, LD1);
  GEMM(Y2h, LD2, wO3, LD2, o3b, Y3, LD2, nullptr, 0, B_ * H_, N2, LD2);
  {
    long long n = (long long)B_ * T_ * H_;
    k_out_y<<<dim3((unsigned)((n + 255) / 256)), dim3(256), 0, stream>>>(y_out, Y3);
  }
}
